// RelNet_34540126994764
// MI455X (gfx1250) — matrix operand order repaired and checked by reading, not compiled, not run on hardware
//
#include <hip/hip_runtime.h>

typedef __attribute__((ext_vector_type(16))) _Float16 v16h;
typedef __attribute__((ext_vector_type(2)))  _Float16 h2;
typedef __attribute__((ext_vector_type(8)))  float    v8f;

// Problem constants: N=512, INPUT_DIM=128, HIDDEN_DIM=256, REL_DIM=128, OUT_DIM=64
// pairs = 512*512 = 262144 ; 16-pair tiles = 16384 ; waves = 2048 ; tiles/wave = 8

// ---------------------------------------------------------------------------
// Kernel 1: Af16[m,c] = f16(b1[c] + sum_k nf[m,k]*W1[k,c])        (k < 128)
//           Bf16[m,c] = f16(       sum_k nf[m,k]*W1[128+k,c])
// ---------------------------------------------------------------------------
__global__ void relnet_prep(const float* __restrict__ nf, const float* __restrict__ W1,
                            const float* __restrict__ b1,
                            _Float16* __restrict__ Af16, _Float16* __restrict__ Bf16) {
  const int m     = blockIdx.x >> 1;
  const int which = blockIdx.x & 1;
  const int c     = threadIdx.x;                 // 0..255
  const float* w    = W1 + which * (128 * 256);
  const float* nrow = nf + m * 128;
  float s = which ? 0.0f : b1[c];
#pragma unroll 4
  for (int k = 0; k < 128; ++k) s = fmaf(nrow[k], w[k * 256 + c], s);
  (which ? Bf16 : Af16)[m * 256 + c] = (_Float16)s;
}

// ---------------------------------------------------------------------------
// Kernel 2: W2 -> fp16 (row-major [256][128]) and zero the 128-float accumulator
// ---------------------------------------------------------------------------
__global__ void relnet_w2cvt(const float* __restrict__ W2, _Float16* __restrict__ W2h,
                             float* __restrict__ accG) {
  const int idx = blockIdx.x * 256 + threadIdx.x;
  if (idx < 256 * 128) W2h[idx] = (_Float16)W2[idx];
  if (idx < 128) accG[idx] = 0.0f;
}

// ---------------------------------------------------------------------------
// Kernel 3 (heavy): per-pair GEMM  r = relu(h @ W2 + b2), summed over pairs.
//   M = 262144 pairs, K = 256, N = 128, fp16 WMMA 16x16x32, f32 accumulate.
// Each wave: 8 tiles of 16 pairs (same i; 16 consecutive j). W2 staged in LDS
// pre-swizzled into the WMMA B-fragment layout (32B contiguous per lane/frag).
// ---------------------------------------------------------------------------
__global__ void __launch_bounds__(256)
relnet_pair(const _Float16* __restrict__ Af16, const _Float16* __restrict__ Bf16,
            const unsigned short* __restrict__ W2h, const float* __restrict__ b2,
            float* __restrict__ accG) {
  extern __shared__ unsigned int lds[];          // 16384 u32 = 64KB of W2 fragments
  const int tid  = threadIdx.x;
  const int lane = tid & 31;
  const int wave = tid >> 5;

  // ---- stage W2 fragments into LDS (layout-permuting copy) ----
  for (int u = tid; u < 16384; u += 256) {
    const int v  = u & 7;
    const int ln = (u >> 3) & 31;
    const int f  = u >> 8;                       // f = kt*8 + nt
    const int nt = f & 7;
    const int kt = f >> 3;
    const int n  = nt * 16 + (ln & 15);
    const int k  = kt * 32 + ((v & 4) << 2) + ((ln >> 4) << 3) + (v & 3) * 2;
    const unsigned int lo = W2h[k * 128 + n];
    const unsigned int hi = W2h[(k + 1) * 128 + n];
    lds[u] = lo | (hi << 16);
  }
  __syncthreads();

  const int gw     = blockIdx.x * 8 + wave;      // global wave id, 0..2047
  const int i      = gw >> 2;                    // left-node index (constant per wave)
  const int row    = lane & 15;                  // M row within tile
  const int laneHi = lane >> 4;

  const h2* Arow = (const h2*)(Af16 + i * 256);

  float b2v[8];
#pragma unroll
  for (int nt = 0; nt < 8; ++nt) b2v[nt] = b2[nt * 16 + row];

  float wsum[8];
#pragma unroll
  for (int nt = 0; nt < 8; ++nt) wsum[nt] = 0.0f;

  const h2 zero2 = {(_Float16)0.0f, (_Float16)0.0f};

#pragma unroll 1
  for (int tt = 0; tt < 8; ++tt) {
    // Opaque zero derived from tt: breaks loop-invariance of the LDS fragment
    // addresses so LICM cannot hoist all 64 fragment loads (which previously
    // ballooned live VGPRs past 500 and spilled every B-fragment to scratch).
    unsigned int z;
    asm("v_and_b32 %0, 0, %1" : "=v"(z) : "v"(tt));
    const unsigned int* ldsT = lds + z;          // == lds at runtime

    const int t  = gw * 8 + tt;                  // tile id
    const int j  = (t & 31) * 16 + row;          // right-node index for this M row
    const h2* Brow = (const h2*)(Bf16 + j * 256);

    v8f accD[8] = {};                            // fresh f32 accumulators per tile

#pragma unroll
    for (int kt = 0; kt < 8; ++kt) {
      // build A fragment: h = relu(A[i,k] + B[j,k]) with packed fp16 math
      union { v16h v; h2 h[8]; } Af;
#pragma unroll
      for (int v = 0; v < 8; ++v) {
        const int kb = kt * 32 + ((v & 4) << 2) + laneHi * 8 + (v & 3) * 2;
        const h2 a2 = Arow[kb >> 1];
        const h2 c2 = Brow[kb >> 1];
        Af.h[v] = __builtin_elementwise_max(a2 + c2, zero2);   // v_pk_add_f16 + pk max
      }
#pragma unroll
      for (int nt = 0; nt < 8; ++nt) {
        union { v16h v; uint4 q[2]; } Bf;
        const uint4* fb = (const uint4*)(ldsT + ((kt * 8 + nt) * 32 + lane) * 8);
        Bf.q[0] = fb[0];
        Bf.q[1] = fb[1];
        accD[nt] = __builtin_amdgcn_wmma_f32_16x16x32_f16(
            false, Af.v, false, Bf.v, (short)0, accD[nt], false, false);
      }
    }

    // epilogue: r = relu(acc + b2); reduce over the 8 M-rows held in this lane
#pragma unroll
    for (int nt = 0; nt < 8; ++nt) {
      float s = 0.0f;
#pragma unroll
      for (int d = 0; d < 8; ++d) s += fmaxf(accD[nt][d] + b2v[nt], 0.0f);
      wsum[nt] += s;
    }
  }

  // merge the two half-wave row sets (lanes L and L^16 hold same n, rows 0-7 vs 8-15)
#pragma unroll
  for (int nt = 0; nt < 8; ++nt) wsum[nt] += __shfl_xor(wsum[nt], 16, 32);

  __syncthreads();                               // done reading W2; reuse LDS for reduce
  float* red = (float*)lds;
  if (lane < 16) {
#pragma unroll
    for (int nt = 0; nt < 8; ++nt) red[wave * 128 + nt * 16 + lane] = wsum[nt];
  }
  __syncthreads();
  if (tid < 128) {
    float s = 0.0f;
#pragma unroll
    for (int w = 0; w < 8; ++w) s += red[w * 128 + tid];
    atomicAdd(accG + tid, s);
  }
}

// ---------------------------------------------------------------------------
// Kernel 4: out = relu((acc/N^2) @ W3 + b3) @ W4 + b4    (1 block, 64 threads)
// ---------------------------------------------------------------------------
__global__ void relnet_head(const float* __restrict__ accG,
                            const float* __restrict__ W3, const float* __restrict__ b3,
                            const float* __restrict__ W4, const float* __restrict__ b4,
                            float* __restrict__ out) {
  __shared__ float m1[64];
  const int t = threadIdx.x;                     // 0..63
  const float inv = 1.0f / (512.0f * 512.0f);
  float s = b3[t];
  for (int k = 0; k < 128; ++k) s = fmaf(accG[k] * inv, W3[k * 64 + t], s);
  m1[t] = fmaxf(s, 0.0f);
  __syncthreads();
  float o = b4[t];
  for (int k = 0; k < 64; ++k) o = fmaf(m1[k], W4[k * 64 + t], o);
  out[t] = o;
}

// ---------------------------------------------------------------------------
extern "C" void kernel_launch(void* const* d_in, const int* in_sizes, int n_in,
                              void* d_out, int out_size, void* d_ws, size_t ws_size,
                              hipStream_t stream) {
  const float* nf = (const float*)d_in[0];
  const float* W1 = (const float*)d_in[1];
  const float* b1 = (const float*)d_in[2];
  const float* W2 = (const float*)d_in[3];
  const float* b2 = (const float*)d_in[4];
  const float* W3 = (const float*)d_in[5];
  const float* b3 = (const float*)d_in[6];
  const float* W4 = (const float*)d_in[7];
  const float* b4 = (const float*)d_in[8];
  float* out = (float*)d_out;

  // workspace layout (bytes): Af16 [512*256 h] | Bf16 [512*256 h] | W2h [256*128 h] | acc [128 f]
  char* ws = (char*)d_ws;
  _Float16* Af16 = (_Float16*)(ws + 0);
  _Float16* Bf16 = (_Float16*)(ws + 262144);
  _Float16* W2h  = (_Float16*)(ws + 524288);
  float*    accG = (float*)(ws + 589824);
  (void)in_sizes; (void)n_in; (void)out_size; (void)ws_size;

  relnet_prep<<<1024, 256, 0, stream>>>(nf, W1, b1, Af16, Bf16);
  relnet_w2cvt<<<128, 256, 0, stream>>>(W2, W2h, accG);
  relnet_pair<<<256, 256, 65536, stream>>>(Af16, Bf16,
                                           (const unsigned short*)W2h, b2, accG);
  relnet_head<<<1, 64, 0, stream>>>(accG, W3, b3, W4, b4, out);
}
